// TemporalRankingLoss_28939489641179
// MI455X (gfx1250) — matrix operand order repaired and checked by reading, not compiled, not run on hardware
//
#include <hip/hip_runtime.h>
#include <hip/hip_bf16.h>

typedef __attribute__((ext_vector_type(16))) __bf16 v16bf;
typedef __attribute__((ext_vector_type(8)))  float  v8f;
typedef unsigned short u16;
typedef unsigned int   u32;

#define NROWS    8192
#define DFEAT    512
#define BT       256        // block tile (BT x BT pairwise tile per workgroup)
#define SAME_EPS 1e-3f

// ---------------- Kernel A: per-row stats + bf16 conversion ----------------
// One wave per row (8 waves / 256-thread block). Computes:
//   dh2[i]  = ||edge_to_i - edge_from_i||^2, or NaN if !is_temporal[i]
//             (NaN makes both strict d_high orderings false == t_pair mask)
//   dl[i]   = ||emb_to_i - emb_from_i||
//   Xb[i,:] = bf16(edge_from_i) (RNE)
//   sqv[i]  = sum(bf16(x)^2) accumulated in f32 (consistent with bf16 Gram)
__global__ __launch_bounds__(256)
void prep_kernel(const float* __restrict__ edge_to, const float* __restrict__ edge_from,
                 const float* __restrict__ emb_to,  const float* __restrict__ emb_from,
                 const unsigned char* __restrict__ tflag,
                 u16* __restrict__ Xb, float* __restrict__ dh2, float* __restrict__ dl,
                 float* __restrict__ sqv, float* lossAcc, u32* cntAcc)
{
  if (blockIdx.x == 0 && threadIdx.x == 0) { *lossAcc = 0.f; *cntAcc = 0u; }
  const int wave = threadIdx.x >> 5;
  const int lane = threadIdx.x & 31;
  const int row  = blockIdx.x * 8 + wave;
  if (row >= NROWS) return;

  const float* to = edge_to   + (size_t)row * DFEAT;
  const float* fr = edge_from + (size_t)row * DFEAT;
  float a = 0.f, s = 0.f;
  #pragma unroll 4
  for (int k = lane; k < DFEAT; k += 32) {
    const float t = to[k], f = fr[k];
    const float e = t - f;
    a += e * e;
    // f32 -> bf16 round-to-nearest-even
    const u32 u = __float_as_uint(f);
    const u16 b = (u16)((u + 0x7FFFu + ((u >> 16) & 1u)) >> 16);
    Xb[(size_t)row * DFEAT + k] = b;
    const float fb = __uint_as_float((u32)b << 16);
    s += fb * fb;
  }
  #pragma unroll
  for (int off = 16; off; off >>= 1) {
    a += __shfl_xor(a, off, 32);
    s += __shfl_xor(s, off, 32);
  }
  if (lane == 0) {
    const bool t = (tflag[row] != 0);
    dh2[row] = t ? a : __uint_as_float(0x7FC00000u);   // NaN disables both orderings
    sqv[row] = s;
    const float e0 = emb_to[2 * row]     - emb_from[2 * row];
    const float e1 = emb_to[2 * row + 1] - emb_from[2 * row + 1];
    dl[row] = sqrtf(e0 * e0 + e1 * e1);
  }
}

// ---------------- Kernel B: fused triangular Gram + masked rank loss ----------------
// Block = 512 threads = 16 waves -> 256x256 output tile (upper triangle of blocks).
// Wave (wi, wj in 0..3) computes a 64x64 subtile = 4x4 WMMA f32 accumulators.
__global__ __launch_bounds__(512)
void pair_kernel(const u16* __restrict__ Xb,
                 const float* __restrict__ dh2, const float* __restrict__ dl,
                 const float* __restrict__ sqv,
                 float* lossAcc, u32* cntAcc)
{
  const int ib = blockIdx.x, jb = blockIdx.y;
  if (jb < ib) return;  // uniform per-block: triangular launch

  __shared__ float s_dh2i[BT], s_dli[BT], s_sqi[BT];
  __shared__ float s_dh2j[BT], s_dlj[BT], s_sqj[BT];
  __shared__ float s_wloss[16];
  __shared__ u32   s_wcnt[16];

  const int tid = threadIdx.x;
  const int iBase = ib * BT, jBase = jb * BT;
  if (tid < BT) {
    const int i = iBase + tid;
    s_dh2i[tid] = dh2[i]; s_dli[tid] = dl[i]; s_sqi[tid] = sqv[i];
  } else {
    const int t = tid - BT, j = jBase + t;
    s_dh2j[t] = dh2[j]; s_dlj[t] = dl[j]; s_sqj[t] = sqv[j];
  }
  __syncthreads();

  const int wave = tid >> 5, lane = tid & 31;
  const int wi = wave & 3, wj = wave >> 2;
  const int rowOff = wi * 64;            // within-tile row offset of this wave
  const int colOff = wj * 64;            // within-tile col offset of this wave
  const int r = lane & 15, half = lane >> 4;

  v8f acc[4][4] = {};

  // A-fragment (16x32 bf16): lane r holds row M=r; half 0 -> K {0..7,16..23},
  //   half 1 -> K {8..15,24..31}  => two 16B loads (+0, +32B).
  // B-fragment (32x16 bf16): lane r holds col N=r; half 0 -> K {0..7,16..23},
  //   half 1 -> K {8..15,24..31}  => two 16B loads (+0, +32B).
  const u16* __restrict__ Abase = Xb + (size_t)(iBase + rowOff + r) * DFEAT + half * 8;
  const u16* __restrict__ Bbase = Xb + (size_t)(jBase + colOff + r) * DFEAT + half * 8;

  for (int kc = 0; kc < DFEAT; kc += 32) {
    v16bf afr[4], bfr[4];
    #pragma unroll
    for (int m = 0; m < 4; ++m) {
      const u16* p = Abase + (size_t)(m * 16) * DFEAT + kc;
      union { v16bf v; uint4 q[2]; } u;
      u.q[0] = *(const uint4*)(p);
      u.q[1] = *(const uint4*)(p + 16);
      afr[m] = u.v;
    }
    #pragma unroll
    for (int n = 0; n < 4; ++n) {
      const u16* p = Bbase + (size_t)(n * 16) * DFEAT + kc;
      union { v16bf v; uint4 q[2]; } u;
      u.q[0] = *(const uint4*)(p);
      u.q[1] = *(const uint4*)(p + 16);
      bfr[n] = u.v;
    }
    #pragma unroll
    for (int m = 0; m < 4; ++m)
      #pragma unroll
      for (int n = 0; n < 4; ++n)
        acc[m][n] = __builtin_amdgcn_wmma_f32_16x16x32_bf16(
            false, afr[m], false, bfr[n], (short)0, acc[m][n], false, false);
  }

  // Spill accumulators to a fully-unrolled scalar view for the epilogue.
  float cf[4][4][8];
  #pragma unroll
  for (int m = 0; m < 4; ++m)
    #pragma unroll
    for (int n = 0; n < 4; ++n) {
      union { v8f v; float f[8]; } c; c.v = acc[m][n];
      #pragma unroll
      for (int v = 0; v < 8; ++v) cf[m][n][v] = c.f[v];
    }

  // Branchless fused epilogue. C layout: VGPR v -> M = v + 8*half, lane -> N = r.
  // For each upper-triangle (jg > ig) Gram element evaluate both ordered pairs.
  // Non-temporal rows carry dh2 = NaN so both strict comparisons fail.
  float loss = 0.f;
  u32   cnt  = 0;
  #pragma unroll
  for (int m = 0; m < 4; ++m) {
    #pragma unroll
    for (int v = 0; v < 8; ++v) {
      const int il = rowOff + m * 16 + half * 8 + v;
      const int ig = iBase + il;
      const float dh2_i = s_dh2i[il], dl_i = s_dli[il], sq_i = s_sqi[il];
      #pragma unroll
      for (int n = 0; n < 4; ++n) {
        const int jl = colOff + n * 16 + r;
        const int jg = jBase + jl;
        const float dh2_j = s_dh2j[jl], dl_j = s_dlj[jl], sq_j = s_sqj[jl];
        const float pd2  = sq_i + sq_j - 2.f * cf[m][n][v];
        const bool same  = (pd2 < SAME_EPS) & (jg > ig);
        const bool m1 = same & (dh2_i < dh2_j) & (dl_i >= dl_j);   // pair (i,j)
        const bool m2 = same & (dh2_j < dh2_i) & (dl_j >= dl_i);   // pair (j,i)
        loss += (m1 ? (dl_i - dl_j) : 0.f) + (m2 ? (dl_j - dl_i) : 0.f);
        cnt  += (u32)m1 + (u32)m2;
      }
    }
  }

  #pragma unroll
  for (int off = 16; off; off >>= 1) {
    loss += __shfl_xor(loss, off, 32);
    cnt  += (u32)__shfl_xor((int)cnt, off, 32);
  }
  if (lane == 0) { s_wloss[wave] = loss; s_wcnt[wave] = cnt; }
  __syncthreads();
  if (tid == 0) {
    float L = 0.f; u32 C = 0;
    #pragma unroll
    for (int w = 0; w < 16; ++w) { L += s_wloss[w]; C += s_wcnt[w]; }
    if (C) { atomicAdd(lossAcc, L); atomicAdd(cntAcc, C); }
  }
}

// ---------------- Kernel C: finalize scalar ----------------
__global__ void final_kernel(const float* lossAcc, const u32* cntAcc, float* out)
{
  if (threadIdx.x == 0 && blockIdx.x == 0) {
    const u32 c = *cntAcc;
    out[0] = (c > 0u) ? (*lossAcc / (float)c) : 0.f;
  }
}

extern "C" void kernel_launch(void* const* d_in, const int* in_sizes, int n_in,
                              void* d_out, int out_size, void* d_ws, size_t ws_size,
                              hipStream_t stream)
{
  const float* edge_to   = (const float*)d_in[0];
  const float* edge_from = (const float*)d_in[1];
  const float* emb_to    = (const float*)d_in[2];
  const float* emb_from  = (const float*)d_in[3];
  const unsigned char* is_temporal = (const unsigned char*)d_in[4];  // numpy bool bytes

  char* ws = (char*)d_ws;
  u16* Xb = (u16*)ws;
  size_t off = (size_t)NROWS * DFEAT * sizeof(u16);       // 8 MB bf16 matrix
  float* dh2 = (float*)(ws + off); off += (size_t)NROWS * sizeof(float);
  float* dl  = (float*)(ws + off); off += (size_t)NROWS * sizeof(float);
  float* sqv = (float*)(ws + off); off += (size_t)NROWS * sizeof(float);
  float* lossAcc = (float*)(ws + off); off += sizeof(float);
  u32*   cntAcc  = (u32*)(ws + off);

  prep_kernel<<<NROWS / 8, 256, 0, stream>>>(edge_to, edge_from, emb_to, emb_from,
                                             is_temporal, Xb, dh2, dl, sqv,
                                             lossAcc, cntAcc);

  dim3 grid(NROWS / BT, NROWS / BT);
  pair_kernel<<<grid, 512, 0, stream>>>(Xb, dh2, dl, sqv, lossAcc, cntAcc);

  final_kernel<<<1, 1, 0, stream>>>(lossAcc, cntAcc, (float*)d_out);
}
